// ALTER2Layer_23785528885731
// MI455X (gfx1250) — matrix operand order repaired and checked by reading, not compiled, not run on hardware
//
#include <hip/hip_runtime.h>

// ---------------------------------------------------------------------------
// Problem constants (from reference): B=128, D=1024, H1=512, H2=128
// Outputs (flat, in order): recover [B*D], c2 [B*H2], Jac [B*D*D]
// Jacobian factorization through H2 bottleneck:
//   L_b    = diag(s2_b) W2 diag(s1_b)            [H2,H1]  (elementwise)
//   G2_b   = L_b @ W1                            [H2,D ]  (WMMA f16)
//   R_b    = W4 @ (diag(s3_b) W3)                [D ,H2]  (WMMA f16)
//   Jac_b  = R_b @ G2_b                          [D ,D ]  (WMMA f16, f32 NT out)
// ---------------------------------------------------------------------------

#define NB 128   // batch
#define ND 1024
#define NH1 512
#define NH2 128

typedef __attribute__((ext_vector_type(16))) _Float16 v16h;
typedef __attribute__((ext_vector_type(8)))  _Float16 v8h;
typedef __attribute__((ext_vector_type(8)))  float    v8f;

__device__ __forceinline__ v8f wmma_f16(v16h a, v16h b, v8f c) {
  return __builtin_amdgcn_wmma_f32_16x16x32_f16(
      /*neg_a=*/false, a, /*neg_b=*/false, b,
      /*c_mod=*/(short)0, c, /*reuse_a=*/false, /*reuse_b=*/false);
}

__device__ __forceinline__ v16h cat8(v8h lo, v8h hi) {
  return __builtin_shufflevector(lo, hi, 0, 1, 2, 3, 4, 5, 6, 7,
                                 8, 9, 10, 11, 12, 13, 14, 15);
}

// ---------------------------------------------------------------------------
// Forward fully-connected layer, f32 scalar with LDS tiling.
// out[b,n] = act( sum_k in[b,k]*W[n,k] + bias[n] ),  B fixed = 128.
// grid.x = N/32, block = 256 (8 waves). Each thread: 1 column n, 16 batches.
// ---------------------------------------------------------------------------
__global__ __launch_bounds__(256) void fc_kernel(
    const float* __restrict__ in, const float* __restrict__ W,
    const float* __restrict__ bias, float* __restrict__ act,
    float* __restrict__ sder, int N, int K, int do_sigmoid) {
  __shared__ float xs[128][33];
  __shared__ float ws[32][33];

  const int tid = threadIdx.x;
  const int nl = tid & 31;          // column within tile (== lane)
  const int bg = tid >> 5;          // wave id -> batch group
  const int n0 = blockIdx.x * 32;

  float acc[16];
#pragma unroll
  for (int j = 0; j < 16; ++j) acc[j] = 0.0f;

  for (int kc = 0; kc < K; kc += 32) {
#pragma unroll
    for (int r = 0; r < 4; ++r) {
      int idx = tid + 256 * r;
      ws[idx >> 5][idx & 31] = W[(size_t)(n0 + (idx >> 5)) * K + kc + (idx & 31)];
    }
#pragma unroll
    for (int r = 0; r < 16; ++r) {
      int idx = tid + 256 * r;
      xs[idx >> 5][idx & 31] = in[(size_t)(idx >> 5) * K + kc + (idx & 31)];
    }
    __syncthreads();
#pragma unroll 8
    for (int kk = 0; kk < 32; ++kk) {
      float wv = ws[nl][kk];
#pragma unroll
      for (int j = 0; j < 16; ++j)
        acc[j] = fmaf(xs[bg * 16 + j][kk], wv, acc[j]);
    }
    __syncthreads();
  }

  const float bv = bias[n0 + nl];
#pragma unroll
  for (int j = 0; j < 16; ++j) {
    int b = bg * 16 + j;
    float z = acc[j] + bv;
    if (do_sigmoid) {
      float a = 1.0f / (1.0f + __expf(-z));
      act[(size_t)b * N + n0 + nl] = a;
      sder[(size_t)b * N + n0 + nl] = a * (1.0f - a);
    } else {
      act[(size_t)b * N + n0 + nl] = z;
    }
  }
}

// hL[b][i][k] = s2[b,i] * W2[i,k] * s1[b,k]   (A operand of G2 GEMM)
__global__ __launch_bounds__(256) void prep_L(
    const float* __restrict__ W2, const float* __restrict__ s1,
    const float* __restrict__ s2, _Float16* __restrict__ hL) {
  const int b = blockIdx.y;
  const int idx = blockIdx.x * 256 + threadIdx.x;   // i*H1 + k
  const int i = idx / NH1, k = idx % NH1;
  hL[(size_t)b * NH2 * NH1 + idx] =
      (_Float16)(s2[b * NH2 + i] * W2[idx] * s1[b * NH1 + k]);
}

// hS3W3T[b][i][k] = s3[b,k] * W3[k,i]  (B^T operand of R GEMM), LDS transpose
__global__ __launch_bounds__(256) void prep_S3W3T(
    const float* __restrict__ W3, const float* __restrict__ s3,
    _Float16* __restrict__ hT) {
  __shared__ float tile[32][33];
  const int nTi = NH2 / 32;                       // 4 tiles along i
  const int k0 = (blockIdx.x / nTi) * 32;
  const int i0 = (blockIdx.x % nTi) * 32;
  const int tid = threadIdx.x;
#pragma unroll
  for (int r = 0; r < 4; ++r) {
    int idx = tid + 256 * r;
    tile[idx >> 5][idx & 31] = W3[(size_t)(k0 + (idx >> 5)) * NH2 + i0 + (idx & 31)];
  }
  __syncthreads();
  for (int bb = 0; bb < 8; ++bb) {
    int b = blockIdx.y * 8 + bb;
#pragma unroll
    for (int r = 0; r < 4; ++r) {
      int idx = tid + 256 * r;
      int ii = idx >> 5, kk = idx & 31;
      hT[(size_t)b * NH2 * NH1 + (size_t)(i0 + ii) * NH1 + k0 + kk] =
          (_Float16)(tile[kk][ii] * s3[b * NH1 + k0 + kk]);
    }
  }
}

// hW1T[d][k] = W1[k][d]  (B^T operand of G2 GEMM), LDS transpose
__global__ __launch_bounds__(256) void prep_W1T(
    const float* __restrict__ W1, _Float16* __restrict__ hT) {
  __shared__ float tile[32][33];
  const int nTk = NH1 / 32;                       // 16 tiles along k
  const int d0 = (blockIdx.x / nTk) * 32;
  const int k0 = (blockIdx.x % nTk) * 32;
  const int tid = threadIdx.x;
#pragma unroll
  for (int r = 0; r < 4; ++r) {
    int idx = tid + 256 * r;
    tile[idx >> 5][idx & 31] = W1[(size_t)(k0 + (idx >> 5)) * ND + d0 + (idx & 31)];
  }
  __syncthreads();
#pragma unroll
  for (int r = 0; r < 4; ++r) {
    int idx = tid + 256 * r;
    int dd = idx >> 5, kk = idx & 31;
    hT[(size_t)(d0 + dd) * NH1 + k0 + kk] = (_Float16)tile[kk][dd];
  }
}

// hW4[o][k] = W4[o][k]  (A operand of R GEMM), straight convert
__global__ __launch_bounds__(256) void prep_cvt(
    const float* __restrict__ src, _Float16* __restrict__ dst) {
  int idx = blockIdx.x * 256 + threadIdx.x;
  dst[idx] = (_Float16)src[idx];
}

// ---------------------------------------------------------------------------
// Batched WMMA GEMM: C_b = A_b[M,K] @ B_b[K,N], B stored transposed [N][K].
// Block = 256 threads = 8 waves (4x2), block tile 128(M) x 128(N).
// Wave tile 32(M) x 64(N) via 2x4 v_wmma_f32_16x16x32_f16 accumulators:
// 8 WMMAs per 12 B128 loads -> ~22 FLOP/byte from cache. Exact tiling.
// EPI: 0 = f32 row-major non-temporal (streaming 512MB Jac, spare L2),
//      1 = f16 row-major, 2 = f16 transposed ([n][m], packed 16B stores).
// ---------------------------------------------------------------------------
template <int EPI, int K>
__global__ __launch_bounds__(256) void wmma_gemm(
    const _Float16* __restrict__ A, long strideA, int lda,
    const _Float16* __restrict__ BT, long strideBT, int ldbt,
    void* __restrict__ Cv, long strideC, int ldc) {
  const int bz = blockIdx.z;
  A += (long)bz * strideA;
  BT += (long)bz * strideBT;

  const int tid = threadIdx.x;
  const int lane = tid & 31;
  const int wave = tid >> 5;
  const int m0 = blockIdx.y * 128 + (wave >> 1) * 32;   // 4 waves over M
  const int n0 = blockIdx.x * 128 + (wave & 1) * 64;    // 2 waves over N
  const int lr = lane & 15;
  const int hi = (lane >> 4) & 1;

  // A 16x32 f16 layout: lane<16 holds K {0..7,16..23}; lane>=16 {8..15,24..31}
  const _Float16* Ar0 = A + (long)(m0 + lr) * lda + hi * 8;
  const _Float16* Ar1 = Ar0 + (long)16 * lda;
  // B 32x16 f16 layout: lane<16 holds K {0..7,16..23}; lane>=16 {8..15,24..31}
  const _Float16* Br0 = BT + (long)(n0 + lr) * ldbt + hi * 8;
  const _Float16* Br1 = Br0 + (long)16 * ldbt;
  const _Float16* Br2 = Br0 + (long)32 * ldbt;
  const _Float16* Br3 = Br0 + (long)48 * ldbt;

  v8f acc[2][4];
#pragma unroll
  for (int i = 0; i < 2; ++i)
#pragma unroll
    for (int j = 0; j < 4; ++j) acc[i][j] = (v8f){};

#pragma unroll 4
  for (int k = 0; k < K; k += 32) {
    v16h a0 = cat8(*(const v8h*)(Ar0 + k), *(const v8h*)(Ar0 + k + 16));
    v16h a1 = cat8(*(const v8h*)(Ar1 + k), *(const v8h*)(Ar1 + k + 16));
    v16h b0 = cat8(*(const v8h*)(Br0 + k), *(const v8h*)(Br0 + k + 16));
    v16h b1 = cat8(*(const v8h*)(Br1 + k), *(const v8h*)(Br1 + k + 16));
    v16h b2 = cat8(*(const v8h*)(Br2 + k), *(const v8h*)(Br2 + k + 16));
    v16h b3 = cat8(*(const v8h*)(Br3 + k), *(const v8h*)(Br3 + k + 16));
    acc[0][0] = wmma_f16(a0, b0, acc[0][0]);
    acc[0][1] = wmma_f16(a0, b1, acc[0][1]);
    acc[0][2] = wmma_f16(a0, b2, acc[0][2]);
    acc[0][3] = wmma_f16(a0, b3, acc[0][3]);
    acc[1][0] = wmma_f16(a1, b0, acc[1][0]);
    acc[1][1] = wmma_f16(a1, b1, acc[1][1]);
    acc[1][2] = wmma_f16(a1, b2, acc[1][2]);
    acc[1][3] = wmma_f16(a1, b3, acc[1][3]);
  }

  // D layout: lane holds column n = n0+nt+lr, rows m = m0+mt+hi*8 + r (r=0..7)
  const int rb = hi * 8;
#pragma unroll
  for (int i = 0; i < 2; ++i) {
#pragma unroll
    for (int j = 0; j < 4; ++j) {
      const int m = m0 + i * 16 + rb, n = n0 + j * 16 + lr;
      if (EPI == 0) {
        float* C = (float*)Cv + (long)bz * strideC;
#pragma unroll
        for (int r = 0; r < 8; ++r)
          __builtin_nontemporal_store(acc[i][j][r], &C[(long)(m + r) * ldc + n]);
      } else if (EPI == 1) {
        _Float16* C = (_Float16*)Cv + (long)bz * strideC;
#pragma unroll
        for (int r = 0; r < 8; ++r)
          C[(long)(m + r) * ldc + n] = (_Float16)acc[i][j][r];
      } else {
        _Float16* C = (_Float16*)Cv + (long)bz * strideC;
        v8h h;
#pragma unroll
        for (int r = 0; r < 8; ++r) h[r] = (_Float16)acc[i][j][r];
        *(v8h*)(C + (long)n * ldc + m) = h;   // contiguous 16B per lane
      }
    }
  }
}

// ---------------------------------------------------------------------------
extern "C" void kernel_launch(void* const* d_in, const int* in_sizes, int n_in,
                              void* d_out, int out_size, void* d_ws, size_t ws_size,
                              hipStream_t stream) {
  const float* x  = (const float*)d_in[0];
  const float* W1 = (const float*)d_in[1];
  const float* b1 = (const float*)d_in[2];
  const float* W2 = (const float*)d_in[3];
  const float* b2 = (const float*)d_in[4];
  const float* W3 = (const float*)d_in[5];
  const float* b3 = (const float*)d_in[6];
  const float* W4 = (const float*)d_in[7];
  const float* b4 = (const float*)d_in[8];

  float* out      = (float*)d_out;
  float* recover  = out;                              // [B, D]
  float* c2out    = out + (size_t)NB * ND;            // [B, H2]
  float* jac      = c2out + (size_t)NB * NH2;         // [B, D, D]

  // Workspace carve-up (256B aligned)
  char* ws = (char*)d_ws;
  size_t off = 0;
  auto carve = [&](size_t bytes) -> char* {
    off = (off + 255) & ~(size_t)255;
    char* p = ws + off;
    off += bytes;
    return p;
  };
  float*    c1  = (float*)carve((size_t)NB * NH1 * 4);
  float*    s1  = (float*)carve((size_t)NB * NH1 * 4);
  float*    s2  = (float*)carve((size_t)NB * NH2 * 4);
  float*    c3  = (float*)carve((size_t)NB * NH1 * 4);
  float*    s3  = (float*)carve((size_t)NB * NH1 * 4);
  _Float16* hW1T = (_Float16*)carve((size_t)ND * NH1 * 2);
  _Float16* hW4  = (_Float16*)carve((size_t)ND * NH1 * 2);
  _Float16* hL   = (_Float16*)carve((size_t)NB * NH2 * NH1 * 2);
  _Float16* hS3W3T = (_Float16*)carve((size_t)NB * NH2 * NH1 * 2);
  _Float16* hG2T = (_Float16*)carve((size_t)NB * ND * NH2 * 2);
  _Float16* hR   = (_Float16*)carve((size_t)NB * ND * NH2 * 2);
  (void)ws_size; (void)in_sizes; (void)n_in; (void)out_size;

  const dim3 blk(256);

  // Forward pass (f32)
  fc_kernel<<<dim3(NH1 / 32), blk, 0, stream>>>(x, W1, b1, c1, s1, NH1, ND, 1);
  fc_kernel<<<dim3(NH2 / 32), blk, 0, stream>>>(c1, W2, b2, c2out, s2, NH2, NH1, 1);
  fc_kernel<<<dim3(NH1 / 32), blk, 0, stream>>>(c2out, W3, b3, c3, s3, NH1, NH2, 1);
  fc_kernel<<<dim3(ND / 32),  blk, 0, stream>>>(c3, W4, b4, recover, recover, ND, NH1, 0);

  // f16 operand prep
  prep_W1T<<<dim3((ND / 32) * (NH1 / 32)), blk, 0, stream>>>(W1, hW1T);
  prep_cvt<<<dim3((ND * NH1) / 256), blk, 0, stream>>>(W4, hW4);
  prep_L<<<dim3((NH2 * NH1) / 256, NB), blk, 0, stream>>>(W2, s1, s2, hL);
  prep_S3W3T<<<dim3((NH1 / 32) * (NH2 / 32), NB / 8), blk, 0, stream>>>(W3, s3, hS3W3T);

  // G2_b = L_b @ W1           (M=H2, N=D, K=H1) -> f16 transposed [d][i]
  wmma_gemm<2, NH1><<<dim3(ND / 128, NH2 / 128, NB), blk, 0, stream>>>(
      hL, (long)NH2 * NH1, NH1, hW1T, 0, NH1,
      hG2T, (long)ND * NH2, NH2);
  // R_b = W4 @ (s3.W3)_b      (M=D, N=H2, K=H1) -> f16 row-major [o][i]
  wmma_gemm<1, NH1><<<dim3(NH2 / 128, ND / 128, NB), blk, 0, stream>>>(
      hW4, 0, NH1, hS3W3T, (long)NH2 * NH1, NH1,
      hR, (long)ND * NH2, NH2);
  // Jac_b = R_b @ G2_b        (M=D, N=D, K=H2) -> f32 NT output
  wmma_gemm<0, NH2><<<dim3(ND / 128, ND / 128, NB), blk, 0, stream>>>(
      hR, (long)ND * NH2, NH2, hG2T, (long)ND * NH2, NH2,
      jac, (long)ND * ND, ND);
}
